// Matching_Score_sent_29197187678290
// MI455X (gfx1250) — matrix operand order repaired and checked by reading, not compiled, not run on hardware
//
#include <hip/hip_runtime.h>

typedef __attribute__((ext_vector_type(16))) _Float16 v16h;
typedef __attribute__((ext_vector_type(8)))  _Float16 v8h;
typedef __attribute__((ext_vector_type(8)))  float    v8f;

#define NB    8192          // batch rows
#define DDIM  256           // feature dim
#define GLOG2E 14.426950408889634f   // 10 * log2(e)  -> exp(10*s) = exp2(GLOG2E*s)
#define NBLK  64            // 8192 / 128 tile blocks per dimension

// ds_swizzle_b32 group-of-32 xor patterns: offset = (xor_mask<<10) | 0x1f
// (SWAPX1/2/4/8/16 from the CDNA5 ISA swizzle table). Immediate operand ->
// zero index-setup VALU, unlike generic __shfl_xor (ds_bpermute + mbcnt/cndmask).
template <int IMM>
__device__ __forceinline__ float swz_xor_add(float x) {
    int i = __builtin_bit_cast(int, x);
    i = __builtin_amdgcn_ds_swizzle(i, IMM);
    return x + __builtin_bit_cast(float, i);
}
#define SWZ_X1  0x041f
#define SWZ_X2  0x081f
#define SWZ_X4  0x101f
#define SWZ_X8  0x201f
#define SWZ_X16 0x401f

// ---------------------------------------------------------------------------
// Kernel 1: row-normalize e and v (fp32), emit f16 normalized rows.
// One wave (32 lanes) per row; 8 floats per lane; swizzle-xor reduction.
// ---------------------------------------------------------------------------
__global__ __launch_bounds__(256) void normalize_f16(
    const float* __restrict__ e, const float* __restrict__ v,
    _Float16* __restrict__ eh, _Float16* __restrict__ vh)
{
    const int wave = (blockIdx.x * blockDim.x + threadIdx.x) >> 5;   // 0..16383
    const int lane = threadIdx.x & 31;

    const float* src;
    _Float16*    dst;
    int row;
    if (wave < NB) { src = e; dst = eh; row = wave; }
    else           { src = v; dst = vh; row = wave - NB; }

    const float* p = src + (size_t)row * DDIM;
    float x[8];
    float ss = 0.0f;
#pragma unroll
    for (int t = 0; t < 8; ++t) {
        x[t] = p[lane + 32 * t];
        ss += x[t] * x[t];
    }
    ss = swz_xor_add<SWZ_X16>(ss);
    ss = swz_xor_add<SWZ_X8>(ss);
    ss = swz_xor_add<SWZ_X4>(ss);
    ss = swz_xor_add<SWZ_X2>(ss);
    ss = swz_xor_add<SWZ_X1>(ss);

    const float scale = 1.0f / fmaxf(sqrtf(ss), 1e-8f);

    _Float16* q = dst + (size_t)row * DDIM;
#pragma unroll
    for (int t = 0; t < 8; ++t)
        q[lane + 32 * t] = (_Float16)(x[t] * scale);
}

// ---------------------------------------------------------------------------
// Kernel 2: one 128x128 tile of E = exp(10 * e_n @ v_n^T) per block.
// 8 waves; wave (wm,wn) owns a 32x64 patch = 2x4 WMMA 16x16 tiles.
// K-loop: 8 steps of 32 using V_WMMA_F32_16X16X32_F16.
// Fused exp + row/col partial reductions; partials to workspace (no atomics,
// fixed summation order -> deterministic).
//
// Fragment layouts (CDNA5 ISA 7.12.2, wave32):
//   A 16x32 f16 : lanes g=lane>>4, m=lane&15 hold row M=m,
//                 halves 0..15 = K (16g .. 16g+15) in this code (contiguous, as B)
//   B 32x16 f16 : lanes hold column N=lane&15,
//                 halves 0..15 = K (16g .. 16g+15)  (contiguous here, like A; not the ISA table order)
//   C/D 16x16 f32: VGPR r -> M = 8g + r, N = lane&15
// Since B[k,n] = v_n[n,k], both A and B chunks are K-contiguous in memory.
// ---------------------------------------------------------------------------
__global__ __launch_bounds__(256) void gram_tile(
    const _Float16* __restrict__ eh, const _Float16* __restrict__ vh,
    float* __restrict__ rowp,   // [NBLK][NB]  rowp[jb][i]
    float* __restrict__ colp)   // [NBLK][NB]  colp[ib][j]
{
    __shared__ float lds_row[2][128];   // slab per wn  (deterministic combine)
    __shared__ float lds_col[4][128];   // slab per wm

    const int lane = threadIdx.x & 31;
    const int w    = threadIdx.x >> 5;  // 0..7
    const int wm   = w >> 1;            // 0..3  (M direction)
    const int wn   = w & 1;             // 0..1  (N direction)
    const int ib   = blockIdx.y;
    const int jb   = blockIdx.x;
    const int i0   = ib * 128;
    const int j0   = jb * 128;
    const int g    = lane >> 4;         // half-wave group
    const int mr   = lane & 15;

    v8f acc[2][4];
#pragma unroll
    for (int tm = 0; tm < 2; ++tm)
#pragma unroll
        for (int tn = 0; tn < 4; ++tn)
            acc[tm][tn] = (v8f){0.f,0.f,0.f,0.f,0.f,0.f,0.f,0.f};

    for (int k0 = 0; k0 < DDIM; k0 += 32) {
        v16h a[2], b[4];
#pragma unroll
        for (int tm = 0; tm < 2; ++tm) {
            const _Float16* pa =
                eh + (size_t)(i0 + wm * 32 + tm * 16 + mr) * DDIM + k0 + 16 * g;
            v8h lo = *(const v8h*)(pa);
            v8h hi = *(const v8h*)(pa + 8);
#pragma unroll
            for (int t = 0; t < 8; ++t) { a[tm][t] = lo[t]; a[tm][t + 8] = hi[t]; }
        }
#pragma unroll
        for (int tn = 0; tn < 4; ++tn) {
            const _Float16* pb =
                vh + (size_t)(j0 + wn * 64 + tn * 16 + mr) * DDIM + k0 + 16 * g;
            b[tn] = *(const v16h*)(pb);
        }
#pragma unroll
        for (int tm = 0; tm < 2; ++tm)
#pragma unroll
            for (int tn = 0; tn < 4; ++tn)
                acc[tm][tn] = __builtin_amdgcn_wmma_f32_16x16x32_f16(
                    false, a[tm], false, b[tn],
                    (short)0, acc[tm][tn], false, false);
    }

    // Fused exp(10*s) = exp2(GLOG2E*s).  |arg| <= ~14.5, so the raw
    // v_exp_f32 is exact enough: skip libm's denorm/overflow fixup path.
    v8f ex[2][4];
#pragma unroll
    for (int tm = 0; tm < 2; ++tm)
#pragma unroll
        for (int tn = 0; tn < 4; ++tn)
#pragma unroll
            for (int r = 0; r < 8; ++r)
                ex[tm][tn][r] = __builtin_amdgcn_exp2f(GLOG2E * acc[tm][tn][r]);

    // Row partials: sum over N.  Value (tm,r) sits at local row
    //   wm*32 + tm*16 + 8*g + r  and column (wn*64 + tn*16 + mr).
    // Sum over tn in-registers, then over the 16 'mr' lanes via ds_swizzle.
#pragma unroll
    for (int tm = 0; tm < 2; ++tm)
#pragma unroll
        for (int r = 0; r < 8; ++r) {
            float rv = ex[tm][0][r] + ex[tm][1][r] + ex[tm][2][r] + ex[tm][3][r];
            rv = swz_xor_add<SWZ_X1>(rv);
            rv = swz_xor_add<SWZ_X2>(rv);
            rv = swz_xor_add<SWZ_X4>(rv);
            rv = swz_xor_add<SWZ_X8>(rv);
            if (mr == 0)
                lds_row[wn][wm * 32 + tm * 16 + 8 * g + r] = rv;
        }

    // Col partials: sum over M (tm, r in-registers; g halves via swizzle 16).
#pragma unroll
    for (int tn = 0; tn < 4; ++tn) {
        float cv = 0.0f;
#pragma unroll
        for (int tm = 0; tm < 2; ++tm)
#pragma unroll
            for (int r = 0; r < 8; ++r)
                cv += ex[tm][tn][r];
        cv = swz_xor_add<SWZ_X16>(cv);
        if (g == 0)
            lds_col[wm][wn * 64 + tn * 16 + mr] = cv;
    }

    __syncthreads();

    const int t = threadIdx.x;
    if (t < 128) {
        rowp[(size_t)jb * NB + i0 + t] = lds_row[0][t] + lds_row[1][t];
    } else {
        const int u = t - 128;
        colp[(size_t)ib * NB + j0 + u] =
            lds_col[0][u] + lds_col[1][u] + lds_col[2][u] + lds_col[3][u];
    }
}

// ---------------------------------------------------------------------------
// Kernel 3: final reduction over the 64 block-partials (fixed order).
// out[0..8191]     = batch_sum_score_d (row sums)
// out[8192..16383] = batch_sum_score_q (col sums)
// ---------------------------------------------------------------------------
__global__ __launch_bounds__(256) void reduce_out(
    const float* __restrict__ rowp, const float* __restrict__ colp,
    float* __restrict__ out)
{
    const int i = blockIdx.x * blockDim.x + threadIdx.x;   // 0..16383
    const float* src = (i < NB) ? (rowp + i) : (colp + (i - NB));
    float s = 0.0f;
#pragma unroll
    for (int b = 0; b < NBLK; ++b)
        s += src[(size_t)b * NB];
    out[i] = s;
}

// ---------------------------------------------------------------------------
extern "C" void kernel_launch(void* const* d_in, const int* in_sizes, int n_in,
                              void* d_out, int out_size, void* d_ws, size_t ws_size,
                              hipStream_t stream)
{
    const float* e = (const float*)d_in[0];
    const float* v = (const float*)d_in[1];
    float* out = (float*)d_out;

    // workspace layout: eh (4MB) | vh (4MB) | rowp (2MB) | colp (2MB)
    _Float16* eh   = (_Float16*)d_ws;
    _Float16* vh   = eh + (size_t)NB * DDIM;
    float*    rowp = (float*)(vh + (size_t)NB * DDIM);
    float*    colp = rowp + (size_t)NBLK * NB;

    // 16384 rows total, 8 waves (1 row each) per 256-thread block
    normalize_f16<<<2 * NB / 8, 256, 0, stream>>>(e, v, eh, vh);

    dim3 grid(NBLK, NBLK);   // x = j blocks, y = i blocks
    gram_tile<<<grid, 256, 0, stream>>>(eh, vh, rowp, colp);

    reduce_out<<<(2 * NB) / 256, 256, 0, stream>>>(rowp, colp, out);
}
